// CausalSelfAttention_66872640798897
// MI455X (gfx1250) — matrix operand order repaired and checked by reading, not compiled, not run on hardware
//
#include <hip/hip_runtime.h>
#include <hip/hip_bf16.h>

// Problem constants (match reference)
#define B_  2
#define T_  2048
#define C_  2048
#define H_  16
#define HS_ 128
#define AL_ 10

typedef __attribute__((ext_vector_type(16))) __bf16 v16bf;
typedef __attribute__((ext_vector_type(8)))  float  v8f;

union BF16Frag { v16bf v; uint4 q[2]; };

__device__ __forceinline__ unsigned short f2bf(float f) {
  union { float f; unsigned u; } x; x.f = f;
  unsigned u = x.u;
  return (unsigned short)((u + 0x7FFFu + ((u >> 16) & 1u)) >> 16); // RNE
}
__device__ __forceinline__ float bf2f(unsigned short h) {
  union { unsigned u; float f; } x; x.u = ((unsigned)h) << 16;
  return x.f;
}

// ---------------------------------------------------------------------------
// f32 -> bf16 conversion (grid-stride)
// ---------------------------------------------------------------------------
__global__ void f32_to_bf16_kernel(const float* __restrict__ in,
                                   unsigned short* __restrict__ out, int n) {
  for (int i = blockIdx.x * blockDim.x + threadIdx.x; i < n;
       i += gridDim.x * blockDim.x)
    out[i] = f2bf(in[i]);
}

// ---------------------------------------------------------------------------
// Tiny adapter K/V precompute (f32 math).
// AK: [H][16][HS] row-major; AVt: [H][HS][16] (transposed for P*V B-frags).
// Rows a >= AL are zero-padded.
// ---------------------------------------------------------------------------
__global__ void adapter_kv_kernel(const float* __restrict__ emb,    // [AL, C]
                                  const float* __restrict__ Wattn,  // [3C, C]
                                  unsigned short* __restrict__ AK,
                                  unsigned short* __restrict__ AVt) {
  int idx = blockIdx.x * blockDim.x + threadIdx.x;
  if (idx >= H_ * 16 * HS_ * 2) return;
  int d   = idx & (HS_ - 1);
  int a   = (idx >> 7) & 15;
  int h   = (idx >> 11) & (H_ - 1);
  int isV = idx >> 15;
  float acc = 0.f;
  if (a < AL_) {
    const float* w = Wattn + (size_t)((isV ? 2 * C_ : C_) + h * HS_ + d) * C_;
    const float* e = emb + (size_t)a * C_;
    for (int i = 0; i < C_; ++i) acc += e[i] * w[i];
  }
  if (isV)
    AVt[h * (16 * HS_) + d * 16 + a] = f2bf(acc);   // transposed
  else
    AK[h * (16 * HS_) + a * HS_ + d] = f2bf(acc);
}

// ---------------------------------------------------------------------------
// bf16 WMMA GEMM:  C[M,N] = A[M,K] * B[N,K]^T     (M,N,K multiples of 128/32)
// Block tile 128x128, 8 waves (2x4), wave tile 64x32, K-tile 32 through LDS,
// register double-buffered global->LDS staging.
// ---------------------------------------------------------------------------
template <bool OUT_BF16>
__global__ __launch_bounds__(256) void gemm_bf16_wmma(
    const unsigned short* __restrict__ A,   // [M,K] bf16
    const unsigned short* __restrict__ Bm,  // [N,K] bf16
    void* __restrict__ Cout, int M, int N, int K) {
  __shared__ unsigned short As[128 * 32];
  __shared__ unsigned short Bs[128 * 32];

  const int tid = threadIdx.x;
  const int lane = tid & 31, wid = tid >> 5;
  const int waveM = wid >> 2, waveN = wid & 3;
  const int m0 = blockIdx.x * 128, n0 = blockIdx.y * 128;
  const int hi = lane >> 4, lo = lane & 15;

  uint4 areg[2], breg[2];
  auto ldg = [&](int k0) {
#pragma unroll
    for (int u = 0; u < 2; ++u) {
      int t = tid + u * 256, row = t >> 2, seg = t & 3;  // 128 rows x 4 b128
      areg[u] = *(const uint4*)&A[(size_t)(m0 + row) * K + k0 + seg * 8];
      breg[u] = *(const uint4*)&Bm[(size_t)(n0 + row) * K + k0 + seg * 8];
    }
  };
  auto sts = [&]() {
#pragma unroll
    for (int u = 0; u < 2; ++u) {
      int t = tid + u * 256, row = t >> 2, seg = t & 3;
      *(uint4*)&As[row * 32 + seg * 8] = areg[u];
      *(uint4*)&Bs[row * 32 + seg * 8] = breg[u];
    }
  };

  v8f acc[4][2];
#pragma unroll
  for (int mi = 0; mi < 4; ++mi)
#pragma unroll
    for (int ni = 0; ni < 2; ++ni)
#pragma unroll
      for (int e = 0; e < 8; ++e) acc[mi][ni][e] = 0.0f;

  ldg(0);
  sts();
  __syncthreads();

  for (int k0 = 0; k0 < K; k0 += 32) {
    const bool have_next = (k0 + 32) < K;
    if (have_next) ldg(k0 + 32);                 // overlap with WMMA below

    BF16Frag af[4], bf[2];
#pragma unroll
    for (int mi = 0; mi < 4; ++mi) {             // A: row=lane%16, K chunks
      int row = waveM * 64 + mi * 16 + lo;
      int c = hi * 16;
      af[mi].q[0] = *(const uint4*)&As[row * 32 + c];
      af[mi].q[1] = *(const uint4*)&As[row * 32 + c + 8];
    }
#pragma unroll
    for (int ni = 0; ni < 2; ++ni) {             // B: col=lane%16, 16 contig K
      int col = waveN * 32 + ni * 16 + lo;
      int kb = hi * 16;
      bf[ni].q[0] = *(const uint4*)&Bs[col * 32 + kb];
      bf[ni].q[1] = *(const uint4*)&Bs[col * 32 + kb + 8];
    }
#pragma unroll
    for (int mi = 0; mi < 4; ++mi)
#pragma unroll
      for (int ni = 0; ni < 2; ++ni)
        acc[mi][ni] = __builtin_amdgcn_wmma_f32_16x16x32_bf16(
            false, af[mi].v, false, bf[ni].v, (short)0, acc[mi][ni], false, false);

    __syncthreads();
    if (have_next) sts();
    __syncthreads();
  }

#pragma unroll
  for (int mi = 0; mi < 4; ++mi)
#pragma unroll
    for (int ni = 0; ni < 2; ++ni)
#pragma unroll
      for (int r = 0; r < 8; ++r) {
        int row = m0 + waveM * 64 + mi * 16 + r + 8 * hi;   // M = r + 8*(lane>=16)
        int col = n0 + waveN * 32 + ni * 16 + lo;           // N = lane%16
        if (OUT_BF16)
          ((unsigned short*)Cout)[(size_t)row * N + col] = f2bf(acc[mi][ni][r]);
        else
          ((float*)Cout)[(size_t)row * N + col] = acc[mi][ni][r];
      }
}

// ---------------------------------------------------------------------------
// RoPE + head split: qkv [B*T,3C] bf16 -> Qh/Kh [B,H,T,HS], Vt [B,H,HS,T].
// V is emitted transposed so the attention kernel can stage it with b128 ops.
// ---------------------------------------------------------------------------
__global__ void rope_split_kernel(const unsigned short* __restrict__ qkv,
                                  const float* __restrict__ rope,  // [T,64,2]
                                  unsigned short* __restrict__ Qh,
                                  unsigned short* __restrict__ Kh,
                                  unsigned short* __restrict__ Vt) {
  int idx = blockIdx.x * blockDim.x + threadIdx.x;
  if (idx >= B_ * T_ * H_ * (HS_ / 2)) return;
  int d2 = idx & 63;
  int h  = (idx >> 6) & (H_ - 1);
  int t  = (idx >> 10) & (T_ - 1);
  int b  = idx >> 21;
  size_t row = (size_t)(b * T_ + t) * (3 * C_);
  int col = h * HS_ + 2 * d2;
  float q0 = bf2f(qkv[row + col]),            q1 = bf2f(qkv[row + col + 1]);
  float k0 = bf2f(qkv[row + C_ + col]),       k1 = bf2f(qkv[row + C_ + col + 1]);
  float v0 = bf2f(qkv[row + 2 * C_ + col]),   v1 = bf2f(qkv[row + 2 * C_ + col + 1]);
  float cs = rope[(t * 64 + d2) * 2 + 0],     sn = rope[(t * 64 + d2) * 2 + 1];
  int bh = b * H_ + h;
  size_t o = ((size_t)bh * T_ + t) * HS_ + 2 * d2;
  Qh[o] = f2bf(q0 * cs - q1 * sn); Qh[o + 1] = f2bf(q1 * cs + q0 * sn);
  Kh[o] = f2bf(k0 * cs - k1 * sn); Kh[o + 1] = f2bf(k1 * cs + k0 * sn);
  size_t ov = ((size_t)bh * HS_ + 2 * d2) * T_ + t;        // transposed
  Vt[ov] = f2bf(v0);
  Vt[ov + T_] = f2bf(v1);
}

// ---------------------------------------------------------------------------
// Flash attention + fused prefix-adapter attention.
// Block = 8 waves, each wave owns 16 queries of one (b,h); KV streamed in
// 32-key tiles through LDS with register double-buffering; online softmax in
// f32; all matmuls via WMMA bf16. All fixed-trip loops fully unrolled.
// ---------------------------------------------------------------------------
__global__ __launch_bounds__(256) void flash_attn_kernel(
    const unsigned short* __restrict__ Qh, const unsigned short* __restrict__ Kh,
    const unsigned short* __restrict__ Vt, const unsigned short* __restrict__ AK,
    const unsigned short* __restrict__ AVt, const float* __restrict__ gating,
    unsigned short* __restrict__ Y) {
  __shared__ unsigned short Ks[32 * 128];       // [tok][hs]
  __shared__ unsigned short Vs[128 * 32];       // [hs][tok]
  __shared__ unsigned short Ps[8][16 * 32];     // per-wave P relayout buffer

  const int tid = threadIdx.x, lane = tid & 31, wid = tid >> 5;
  const int hi = lane >> 4, lo = lane & 15;
  const int bh = blockIdx.x, b = bh / H_, h = bh % H_;
  const int q0 = blockIdx.y * 128;
  const int qw = q0 + wid * 16;
  const float scale = 0.08838834764831845f;     // 1/sqrt(128)
  const size_t base  = (size_t)bh * T_ * HS_;   // Qh/Kh: [bh][t][hs]
  const size_t vbase = (size_t)bh * HS_ * T_;   // Vt:    [bh][hs][t]

  uint4 kreg[2], vreg[2];
  auto ldg_tile = [&](int kt) {
#pragma unroll
    for (int u = 0; u < 2; ++u) {
      int t = tid + u * 256;
      int row = t >> 4, seg = t & 15;           // K: 32 rows x 16 b128
      kreg[u] = *(const uint4*)&Kh[base + (size_t)(kt + row) * HS_ + seg * 8];
      int vrow = t >> 2, vseg = t & 3;          // V: 128 hs-rows x 4 b128
      vreg[u] = *(const uint4*)&Vt[vbase + (size_t)vrow * T_ + kt + vseg * 8];
    }
  };
  auto sts_tile = [&]() {
#pragma unroll
    for (int u = 0; u < 2; ++u) {
      int t = tid + u * 256;
      int row = t >> 4, seg = t & 15;
      *(uint4*)&Ks[row * 128 + seg * 8] = kreg[u];
      int vrow = t >> 2, vseg = t & 3;
      *(uint4*)&Vs[vrow * 32 + vseg * 8] = vreg[u];
    }
  };

  BF16Frag qf[4];                               // Q A-fragments (K = 4 x 32)
  {
    const unsigned short* qrow = Qh + base + (size_t)(qw + lo) * HS_;
    int c = hi * 16;
#pragma unroll
    for (int ks = 0; ks < 4; ++ks) {
      qf[ks].q[0] = *(const uint4*)&qrow[ks * 32 + c];
      qf[ks].q[1] = *(const uint4*)&qrow[ks * 32 + c + 8];
    }
  }

  v8f o[8];
#pragma unroll
  for (int n = 0; n < 8; ++n)
#pragma unroll
    for (int e = 0; e < 8; ++e) o[n][e] = 0.f;
  float mrow[8], lrow[8];
#pragma unroll
  for (int r = 0; r < 8; ++r) { mrow[r] = -__builtin_inff(); lrow[r] = 0.f; }

  const int kend = qw + 16;                     // causal bound (exclusive)
  const int ktlim = q0 + 128;

  ldg_tile(0);
  sts_tile();
  __syncthreads();

  for (int kt = 0; kt < ktlim; kt += 32) {
    const bool have_next = (kt + 32) < ktlim;
    if (have_next) ldg_tile(kt + 32);           // overlap with compute below

    if (kt < kend) {
      v8f s[2];
#pragma unroll
      for (int ni = 0; ni < 2; ++ni)
#pragma unroll
        for (int e = 0; e < 8; ++e) s[ni][e] = 0.f;
#pragma unroll
      for (int ni = 0; ni < 2; ++ni)
#pragma unroll
        for (int ks = 0; ks < 4; ++ks) {
          BF16Frag bk;
          int off = (ni * 16 + lo) * 128 + ks * 32 + hi * 16;
          bk.q[0] = *(const uint4*)&Ks[off];
          bk.q[1] = *(const uint4*)&Ks[off + 8];
          s[ni] = __builtin_amdgcn_wmma_f32_16x16x32_bf16(
              false, qf[ks].v, false, bk.v, (short)0, s[ni], false, false);
        }
      // scale + causal mask + online softmax (16-lane row reductions)
#pragma unroll
      for (int r = 0; r < 8; ++r) {
        int qidx = qw + r + 8 * hi;
        float s0 = s[0][r] * scale, s1 = s[1][r] * scale;
        if (kt + lo > qidx)      s0 = -__builtin_inff();
        if (kt + 16 + lo > qidx) s1 = -__builtin_inff();
        float mt = fmaxf(s0, s1);
#pragma unroll
        for (int m = 1; m < 16; m <<= 1) mt = fmaxf(mt, __shfl_xor(mt, m, 32));
        float mnew = fmaxf(mrow[r], mt);
        float corr = __expf(mrow[r] - mnew);
        float p0 = __expf(s0 - mnew), p1 = __expf(s1 - mnew);
        float rs = p0 + p1;
#pragma unroll
        for (int m = 1; m < 16; m <<= 1) rs += __shfl_xor(rs, m, 32);
        lrow[r] = lrow[r] * corr + rs;
        mrow[r] = mnew;
#pragma unroll
        for (int n = 0; n < 8; ++n) o[n][r] *= corr;
        int M = r + 8 * hi;
        Ps[wid][M * 32 + lo]      = f2bf(p0);   // C-layout -> LDS row-major
        Ps[wid][M * 32 + 16 + lo] = f2bf(p1);
      }
      BF16Frag pf;                              // reload as A-fragment
      {
        int c = hi * 16;
        pf.q[0] = *(const uint4*)&Ps[wid][lo * 32 + c];
        pf.q[1] = *(const uint4*)&Ps[wid][lo * 32 + c + 8];
      }
#pragma unroll
      for (int n = 0; n < 8; ++n) {             // O += P * V
        BF16Frag bv;
        int off = (n * 16 + lo) * 32 + hi * 16;
        bv.q[0] = *(const uint4*)&Vs[off];
        bv.q[1] = *(const uint4*)&Vs[off + 8];
        o[n] = __builtin_amdgcn_wmma_f32_16x16x32_bf16(
            false, pf.v, false, bv.v, (short)0, o[n], false, false);
      }
    }
    __syncthreads();
    if (have_next) sts_tile();
    __syncthreads();
  }

  // ---- fused prefix-adapter attention (16-key padded tile, own softmax) ----
#pragma unroll
  for (int i = tid; i < 4096; i += 256) Vs[i] = 0;
#pragma unroll
  for (int i = tid; i < 2048; i += 256) Ks[i] = AK[h * 2048 + i];
  __syncthreads();
#pragma unroll
  for (int i = tid; i < 2048; i += 256) {       // AVt: [hs][16 tok]
    int hs = i >> 4, tok = i & 15;
    Vs[hs * 32 + tok] = AVt[h * 2048 + i];
  }
  __syncthreads();

  v8f sa;
#pragma unroll
  for (int e = 0; e < 8; ++e) sa[e] = 0.f;
#pragma unroll
  for (int ks = 0; ks < 4; ++ks) {
    BF16Frag bk;
    int off = lo * 128 + ks * 32 + hi * 16;
    bk.q[0] = *(const uint4*)&Ks[off];
    bk.q[1] = *(const uint4*)&Ks[off + 8];
    sa = __builtin_amdgcn_wmma_f32_16x16x32_bf16(
        false, qf[ks].v, false, bk.v, (short)0, sa, false, false);
  }
  float la[8];
#pragma unroll
  for (int r = 0; r < 8; ++r) {
    float sv = sa[r] * scale;
    if (lo >= AL_) sv = -__builtin_inff();
    float mt = sv;
#pragma unroll
    for (int m = 1; m < 16; m <<= 1) mt = fmaxf(mt, __shfl_xor(mt, m, 32));
    float p = __expf(sv - mt);
    float rs = p;
#pragma unroll
    for (int m = 1; m < 16; m <<= 1) rs += __shfl_xor(rs, m, 32);
    la[r] = rs;
    int M = r + 8 * hi;
    Ps[wid][M * 32 + lo]      = f2bf(p);
    Ps[wid][M * 32 + 16 + lo] = 0;              // zero-pad keys 16..31
  }
  BF16Frag pf;
  {
    int c = hi * 16;
    pf.q[0] = *(const uint4*)&Ps[wid][lo * 32 + c];
    pf.q[1] = *(const uint4*)&Ps[wid][lo * 32 + c + 8];
  }
  v8f oa[8];
#pragma unroll
  for (int n = 0; n < 8; ++n)
#pragma unroll
    for (int e = 0; e < 8; ++e) oa[n][e] = 0.f;
#pragma unroll
  for (int n = 0; n < 8; ++n) {
    BF16Frag bv;
    int off = (n * 16 + lo) * 32 + hi * 16;
    bv.q[0] = *(const uint4*)&Vs[off];
    bv.q[1] = *(const uint4*)&Vs[off + 8];
    oa[n] = __builtin_amdgcn_wmma_f32_16x16x32_bf16(
        false, pf.v, false, bv.v, (short)0, oa[n], false, false);
  }

  const float g = gating[h];
#pragma unroll
  for (int n = 0; n < 8; ++n) {
#pragma unroll
    for (int r = 0; r < 8; ++r) {
      float val = o[n][r] / lrow[r] + g * (oa[n][r] / la[r]);
      Y[(size_t)(b * T_ + qw + r + 8 * hi) * C_ + h * HS_ + n * 16 + lo] =
          f2bf(val);
    }
  }
}

// ---------------------------------------------------------------------------
extern "C" void kernel_launch(void* const* d_in, const int* in_sizes, int n_in,
                              void* d_out, int out_size, void* d_ws, size_t ws_size,
                              hipStream_t stream) {
  const float* x      = (const float*)d_in[0];
  const float* rope   = (const float*)d_in[1];
  // d_in[2] = mask (bool) — causality computed analytically, unused
  const float* Wattn  = (const float*)d_in[3];
  const float* Wproj  = (const float*)d_in[4];
  const float* aemb   = (const float*)d_in[5];
  const float* gating = (const float*)d_in[6];
  // d_in[7] = max_seq_length — unused
  (void)in_sizes; (void)n_in; (void)out_size; (void)ws_size;

  char* ws = (char*)d_ws;
  size_t off = 0;
  auto alloc = [&](size_t bytes) {
    char* p = ws + off;
    off = (off + bytes + 255) & ~(size_t)255;
    return p;
  };
  unsigned short* xbf   = (unsigned short*)alloc((size_t)B_ * T_ * C_ * 2);
  unsigned short* wabf  = (unsigned short*)alloc((size_t)3 * C_ * C_ * 2);
  unsigned short* wpbf  = (unsigned short*)alloc((size_t)C_ * C_ * 2);
  unsigned short* qkvbf = (unsigned short*)alloc((size_t)B_ * T_ * 3 * C_ * 2);
  unsigned short* Qh    = (unsigned short*)alloc((size_t)B_ * T_ * C_ * 2);
  unsigned short* Kh    = (unsigned short*)alloc((size_t)B_ * T_ * C_ * 2);
  unsigned short* Vt    = (unsigned short*)alloc((size_t)B_ * T_ * C_ * 2);
  unsigned short* Ybf   = (unsigned short*)alloc((size_t)B_ * T_ * C_ * 2);
  unsigned short* AK    = (unsigned short*)alloc((size_t)H_ * 16 * HS_ * 2);
  unsigned short* AVt   = (unsigned short*)alloc((size_t)H_ * 16 * HS_ * 2);

  const int threads = 256;
  f32_to_bf16_kernel<<<2048, threads, 0, stream>>>(x, xbf, B_ * T_ * C_);
  f32_to_bf16_kernel<<<2048, threads, 0, stream>>>(Wattn, wabf, 3 * C_ * C_);
  f32_to_bf16_kernel<<<2048, threads, 0, stream>>>(Wproj, wpbf, C_ * C_);
  adapter_kv_kernel<<<(H_ * 16 * HS_ * 2 + 255) / 256, threads, 0, stream>>>(
      aemb, Wattn, AK, AVt);

  dim3 g1(B_ * T_ / 128, 3 * C_ / 128);
  gemm_bf16_wmma<true><<<g1, threads, 0, stream>>>(xbf, wabf, qkvbf,
                                                   B_ * T_, 3 * C_, C_);
  rope_split_kernel<<<(B_ * T_ * H_ * (HS_ / 2) + 255) / 256, threads, 0,
                      stream>>>(qkvbf, rope, Qh, Kh, Vt);

  dim3 g2(B_ * H_, T_ / 128);
  flash_attn_kernel<<<g2, threads, 0, stream>>>(Qh, Kh, Vt, AK, AVt, gating, Ybf);

  dim3 g3(B_ * T_ / 128, C_ / 128);
  gemm_bf16_wmma<false><<<g3, threads, 0, stream>>>(Ybf, wpbf, d_out,
                                                    B_ * T_, C_, C_);
}
